// MultiHeadAttention_72619307041469
// MI455X (gfx1250) — matrix operand order repaired and checked by reading, not compiled, not run on hardware
//
#include <hip/hip_runtime.h>

// Problem constants (B=1)
constexpr int Sdim = 2048;   // sequence length
constexpr int Vdim = 1024;   // model dim
constexpr int Hn   = 16;     // heads
constexpr int Dh   = 64;     // head dim

typedef __attribute__((ext_vector_type(16))) _Float16 v16h;
typedef __attribute__((ext_vector_type(8)))  _Float16 v8h;
typedef __attribute__((ext_vector_type(8)))  float    v8f;

// ---------------------------------------------------------------------------
// CDNA5 async copy: per-lane 16B global -> LDS, tracked by ASYNCcnt
// (CDNA5 ISA). One instruction moves 512B per wave.
// ---------------------------------------------------------------------------
__device__ __forceinline__ void async_copy_b128(unsigned lds_off, const void* gaddr) {
  asm volatile("global_load_async_to_lds_b128 %0, %1, off"
               :: "v"(lds_off), "v"(gaddr)
               : "memory");
}
__device__ __forceinline__ void wait_asynccnt0() {
  asm volatile("s_wait_asynccnt 0" ::: "memory");
}

// ---------------------------------------------------------------------------
// WMMA fragment loaders, for CDNA5 wave32; A and B share a contiguous K order (not the ISA table order)
//
// A (16x32 f16, row-major MxK): lane&15 = M row; lane>>4 selects
//   K in {half*16 .. half*16+15} contiguous
// B (32x16 f16, fed row-major N x K): lane&15 = N col; lane>>4 selects
//   K in {half*16 .. half*16+15} contiguous
// C/D (16x16 f32): VGPR r -> M = half*8 + r, N = lane&15
// ---------------------------------------------------------------------------
__device__ __forceinline__ v16h frag_a(const _Float16* p /* row base + k0 + half*16 */) {
  v8h lo = *(const v8h*)(p);
  v8h hi = *(const v8h*)(p + 8);
  return __builtin_shufflevector(lo, hi, 0,1,2,3,4,5,6,7,8,9,10,11,12,13,14,15);
}
__device__ __forceinline__ v16h frag_b(const _Float16* p /* row base + k0 + half*16 */) {
  v8h lo = *(const v8h*)(p);
  v8h hi = *(const v8h*)(p + 8);
  return __builtin_shufflevector(lo, hi, 0,1,2,3,4,5,6,7,8,9,10,11,12,13,14,15);
}

// ---------------------------------------------------------------------------
// LDS-tiled, double-buffered NT GEMM: C[M,N] (f32) = A[M,K] f16 * B[N,K]^T f16
// Block = 256 threads (8 waves) computes a 128x64 C tile.
// Pipeline per 32-wide K step:
//   s_wait_asynccnt 0  (current buffer landed, this wave)
//   barrier            (current buffer landed, all waves; prev compute done)
//   issue async fills for NEXT buffer          <-- overlaps with compute
//   10x ds_load_b128 (A frag + 4 B frags), 4x v_wmma back-to-back
// LDS: 2*(8KB A + 4KB B) = 24KB. Grid: (N/64, M/128); M%128,N%64,K%32 == 0.
// ---------------------------------------------------------------------------
__global__ __launch_bounds__(256) void gemm_nt_lds(const _Float16* __restrict__ A,
                                                   const _Float16* __restrict__ B,
                                                   float* __restrict__ C,
                                                   int M, int N, int K,
                                                   int lda, int ldb, int ldc) {
  __shared__ _Float16 smA[2][128 * 32];   // 2 x 8 KB
  __shared__ _Float16 smB[2][64 * 32];    // 2 x 4 KB

  const int tid     = threadIdx.x;
  const int lane    = tid & 31;
  const int wave    = tid >> 5;
  const int halfsel = lane >> 4;
  const int ncol    = lane & 15;
  const int mblk    = blockIdx.y * 128;
  const int nblk    = blockIdx.x * 64;

  // low 32 bits of the flat LDS address == LDS byte offset
  const unsigned ldsA[2] = { (unsigned)(uintptr_t)&smA[0][0],
                             (unsigned)(uintptr_t)&smA[1][0] };
  const unsigned ldsB[2] = { (unsigned)(uintptr_t)&smB[0][0],
                             (unsigned)(uintptr_t)&smB[1][0] };

  // Per-thread async-copy assignment (16B chunks; 4 chunks per 32-half row):
  //   A: 512 chunks -> chunks {tid, tid+256};  B: 256 chunks -> chunk {tid}
  const int ra0 = tid >> 2;
  const int ra1 = (tid + 256) >> 2;
  const int rb  = tid >> 2;
  const int seg = tid & 3;                 // (tid+256)&3 == tid&3

  auto issue_copies = [&](int buf, int k0) {
    async_copy_b128(ldsA[buf] + (unsigned)tid * 16,
                    A + (size_t)(mblk + ra0) * lda + k0 + seg * 8);
    async_copy_b128(ldsA[buf] + (unsigned)(tid + 256) * 16,
                    A + (size_t)(mblk + ra1) * lda + k0 + seg * 8);
    async_copy_b128(ldsB[buf] + (unsigned)tid * 16,
                    B + (size_t)(nblk + rb) * ldb + k0 + seg * 8);
  };

  v8f zero = {};
  v8f acc[4];
#pragma unroll
  for (int t = 0; t < 4; ++t) acc[t] = zero;

  issue_copies(0, 0);                      // prologue fill of buffer 0
  int buf = 0;
  for (int k0 = 0; k0 < K; k0 += 32) {
    wait_asynccnt0();                      // this wave's fills of `buf` landed
    __syncthreads();                       // all waves' fills landed; prev compute done
    if (k0 + 32 < K) issue_copies(buf ^ 1, k0 + 32);  // overlap next fill

    v16h a = frag_a(&smA[buf][(wave * 16 + ncol) * 32 + halfsel * 16]);
    v16h bf[4];
#pragma unroll
    for (int t = 0; t < 4; ++t)
      bf[t] = frag_b(&smB[buf][(t * 16 + ncol) * 32 + halfsel * 16]);
#pragma unroll
    for (int t = 0; t < 4; ++t)
      acc[t] = __builtin_amdgcn_wmma_f32_16x16x32_f16(
          false, a, false, bf[t], (short)0, acc[t], false, false);

    buf ^= 1;
  }

  const int m0 = mblk + wave * 16;
#pragma unroll
  for (int t = 0; t < 4; ++t) {
    const int n = nblk + t * 16 + ncol;
#pragma unroll
    for (int r = 0; r < 8; ++r) {
      const int m = m0 + halfsel * 8 + r;
      C[(size_t)m * ldc + n] = acc[t][r];
    }
  }
}

// ---------------------------------------------------------------------------
// Fused score kernel (per head): dots = Qu[S,D] * Kh[S,D]^T via WMMA, then
//   scores[i,j] = (j > i) ? -inf
//               : (dots[i,j] + P[i, (j-i-1) & (S-1)]) * 1/sqrt(D)
// Operands are small & L2-resident; direct-global fragment loads.
// Grid: (S/64, S/128).
// ---------------------------------------------------------------------------
__global__ __launch_bounds__(256) void score_kernel(const _Float16* __restrict__ Qu,
                                                    const _Float16* __restrict__ Kh,
                                                    const float* __restrict__ P,
                                                    float* __restrict__ scores) {
  const int lane    = threadIdx.x & 31;
  const int wave    = threadIdx.x >> 5;
  const int halfsel = lane >> 4;
  const int m0      = (blockIdx.y * 8 + wave) * 16;
  const int n0      = blockIdx.x * 64;
  const int mrow    = m0 + (lane & 15);
  const int ncol    = lane & 15;

  v8f zero = {};
  v8f acc[4];
#pragma unroll
  for (int t = 0; t < 4; ++t) acc[t] = zero;

#pragma unroll
  for (int k0 = 0; k0 < Dh; k0 += 32) {
    v16h a = frag_a(Qu + (size_t)mrow * Dh + k0 + halfsel * 16);
    v16h bf[4];
#pragma unroll
    for (int t = 0; t < 4; ++t)
      bf[t] = frag_b(Kh + (size_t)(n0 + t * 16 + ncol) * Dh + k0 + halfsel * 16);
#pragma unroll
    for (int t = 0; t < 4; ++t)
      acc[t] = __builtin_amdgcn_wmma_f32_16x16x32_f16(
          false, a, false, bf[t], (short)0, acc[t], false, false);
  }

  const float scale = 0.125f;                 // 1/sqrt(64)
  const float ninf  = -__builtin_inff();
#pragma unroll
  for (int t = 0; t < 4; ++t) {
    const int j = n0 + t * 16 + ncol;
#pragma unroll
    for (int r = 0; r < 8; ++r) {
      const int i = m0 + halfsel * 8 + r;
      float val;
      if (j > i) {
        val = ninf;                              // causal mask
      } else {
        const int c = (j - i - 1) & (Sdim - 1);  // Transformer-XL shift gather
        val = (acc[t][r] + P[(size_t)i * Sdim + c]) * scale;
      }
      scores[(size_t)i * Sdim + j] = val;
    }
  }
}

// ---------------------------------------------------------------------------
// Row softmax: scores f32 [S,S] -> probs f16 [S,S]. One block per row.
// ---------------------------------------------------------------------------
__global__ __launch_bounds__(256) void softmax_kernel(const float* __restrict__ scores,
                                                      _Float16* __restrict__ probs) {
  __shared__ float red[256];
  const int row = blockIdx.x;
  const int tid = threadIdx.x;
  const float* p = scores + (size_t)row * Sdim;

  float mx = -__builtin_inff();
  for (int j = tid; j < Sdim; j += 256) mx = fmaxf(mx, p[j]);
  red[tid] = mx;
  __syncthreads();
  for (int s = 128; s > 0; s >>= 1) {
    if (tid < s) red[tid] = fmaxf(red[tid], red[tid + s]);
    __syncthreads();
  }
  mx = red[0];
  __syncthreads();

  float sum = 0.f;
  for (int j = tid; j < Sdim; j += 256) sum += __expf(p[j] - mx);
  red[tid] = sum;
  __syncthreads();
  for (int s = 128; s > 0; s >>= 1) {
    if (tid < s) red[tid] += red[tid + s];
    __syncthreads();
  }
  const float inv = 1.f / red[0];

  _Float16* q = probs + (size_t)row * Sdim;
  for (int j = tid; j < Sdim; j += 256)
    q[j] = (_Float16)(__expf(p[j] - mx) * inv);
}

// ---------------------------------------------------------------------------
// Elementwise f32 -> f16
// ---------------------------------------------------------------------------
__global__ __launch_bounds__(256) void cvt_f32_f16(const float* __restrict__ src,
                                                   _Float16* __restrict__ dst, int n) {
  int i = blockIdx.x * 256 + threadIdx.x;
  if (i < n) dst[i] = (_Float16)src[i];
}

// ---------------------------------------------------------------------------
// Split qkv (f32 [S, 3V]) into per-head f16 tensors:
//   qu[h][i][d] = q + u_vec,  qv[h][i][d] = q + v_vec,  k[h][i][d],
//   vT[h][d][i] (transposed so P*V becomes an NT GEMM)
// ---------------------------------------------------------------------------
__global__ __launch_bounds__(256) void pack_qkv(const float* __restrict__ qkv,
                                                const float* __restrict__ u,
                                                const float* __restrict__ vvec,
                                                _Float16* __restrict__ qu,
                                                _Float16* __restrict__ qv,
                                                _Float16* __restrict__ kh,
                                                _Float16* __restrict__ vT) {
  int idx = blockIdx.x * 256 + threadIdx.x;
  if (idx >= Sdim * Vdim) return;
  const int i = idx / Vdim;
  const int c = idx % Vdim;
  const int h = c / Dh;
  const int d = c % Dh;
  const size_t row = (size_t)i * (3 * Vdim);
  const float q = qkv[row + c];
  const float k = qkv[row + Vdim + c];
  const float v = qkv[row + 2 * Vdim + c];
  const size_t hd = ((size_t)h * Sdim + i) * Dh + d;
  qu[hd] = (_Float16)(q + u[c]);
  qv[hd] = (_Float16)(q + vvec[c]);
  kh[hd] = (_Float16)k;
  vT[((size_t)h * Dh + d) * Sdim + i] = (_Float16)v;
}

// RP f32 [S,V] -> per-head f16 [H][S][D]
__global__ __launch_bounds__(256) void pack_rp(const float* __restrict__ RP,
                                               _Float16* __restrict__ rp) {
  int idx = blockIdx.x * 256 + threadIdx.x;
  if (idx >= Sdim * Vdim) return;
  const int j = idx / Vdim;
  const int c = idx % Vdim;
  const int h = c / Dh;
  const int d = c % Dh;
  rp[((size_t)h * Sdim + j) * Dh + d] = (_Float16)RP[(size_t)j * Vdim + c];
}

// ---------------------------------------------------------------------------
extern "C" void kernel_launch(void* const* d_in, const int* in_sizes, int n_in,
                              void* d_out, int out_size, void* d_ws, size_t ws_size,
                              hipStream_t stream) {
  const float* x    = (const float*)d_in[0];
  const float* Wqkv = (const float*)d_in[1];
  const float* Wout = (const float*)d_in[2];
  const float* Wpos = (const float*)d_in[3];
  const float* u    = (const float*)d_in[4];
  const float* vv   = (const float*)d_in[5];
  const float* rel  = (const float*)d_in[6];
  float* out = (float*)d_out;

  // workspace carve-up (256B aligned)
  char* base = (char*)d_ws;
  size_t off = 0;
  auto alloc = [&](size_t bytes) -> void* {
    void* p = base + off;
    off = (off + bytes + (size_t)255) & ~(size_t)255;
    return p;
  };
  const size_t SV = (size_t)Sdim * Vdim;
  const size_t SS = (size_t)Sdim * Sdim;

  _Float16* xh     = (_Float16*)alloc(SV * 2);
  _Float16* Wqkvh  = (_Float16*)alloc((size_t)3 * Vdim * Vdim * 2);
  _Float16* Wposh  = (_Float16*)alloc((size_t)Vdim * Vdim * 2);
  _Float16* Wouth  = (_Float16*)alloc((size_t)Vdim * Vdim * 2);
  _Float16* relh   = (_Float16*)alloc(SV * 2);
  float*    qkv32  = (float*)   alloc(SV * 3 * 4);
  float*    RP32   = (float*)   alloc(SV * 4);
  _Float16* quh    = (_Float16*)alloc(SV * 2);   // [H][S][D]
  _Float16* qvh    = (_Float16*)alloc(SV * 2);   // [H][S][D]
  _Float16* khh    = (_Float16*)alloc(SV * 2);   // [H][S][D]
  _Float16* vTh    = (_Float16*)alloc(SV * 2);   // [H][D][S]
  _Float16* rph    = (_Float16*)alloc(SV * 2);   // [H][S][D]
  float*    Pbuf   = (float*)   alloc(SS * 4);   // per-head, reused
  float*    scores = (float*)   alloc(SS * 4);   // per-head, reused
  _Float16* probs  = (_Float16*)alloc(SS * 2);   // per-head, reused
  float*    att32  = (float*)   alloc(SV * 4);
  _Float16* atth   = (_Float16*)alloc(SV * 2);
  (void)ws_size; (void)in_sizes; (void)n_in; (void)out_size;

  const int blk = 256;
  auto gblk = [](size_t n) { return (int)((n + 255) / 256); };

  // 1) convert inputs to f16
  cvt_f32_f16<<<gblk(SV), blk, 0, stream>>>(x, xh, (int)SV);
  cvt_f32_f16<<<gblk((size_t)3 * Vdim * Vdim), blk, 0, stream>>>(Wqkv, Wqkvh, 3 * Vdim * Vdim);
  cvt_f32_f16<<<gblk((size_t)Vdim * Vdim), blk, 0, stream>>>(Wpos, Wposh, Vdim * Vdim);
  cvt_f32_f16<<<gblk((size_t)Vdim * Vdim), blk, 0, stream>>>(Wout, Wouth, Vdim * Vdim);
  cvt_f32_f16<<<gblk(SV), blk, 0, stream>>>(rel, relh, (int)SV);

  // 2) qkv = x @ Wqkv^T   [2048 x 3072 x 1024]
  gemm_nt_lds<<<dim3(3 * Vdim / 64, Sdim / 128), blk, 0, stream>>>(
      xh, Wqkvh, qkv32, Sdim, 3 * Vdim, Vdim, Vdim, Vdim, 3 * Vdim);
  // 3) RP = rel_embed @ Wpos^T   [2048 x 1024 x 1024]
  gemm_nt_lds<<<dim3(Vdim / 64, Sdim / 128), blk, 0, stream>>>(
      relh, Wposh, RP32, Sdim, Vdim, Vdim, Vdim, Vdim, Vdim);

  // 4) head packing
  pack_qkv<<<gblk(SV), blk, 0, stream>>>(qkv32, u, vv, quh, qvh, khh, vTh);
  pack_rp<<<gblk(SV), blk, 0, stream>>>(RP32, rph);

  // 5) per-head attention (working set stays L2-resident)
  for (int h = 0; h < Hn; ++h) {
    const _Float16* qu_h = quh + (size_t)h * Sdim * Dh;
    const _Float16* qv_h = qvh + (size_t)h * Sdim * Dh;
    const _Float16* k_h  = khh + (size_t)h * Sdim * Dh;
    const _Float16* vT_h = vTh + (size_t)h * Dh * Sdim;
    const _Float16* rp_h = rph + (size_t)h * Sdim * Dh;

    // P = (q + v_vec) @ RPh^T   [S x S x D]
    gemm_nt_lds<<<dim3(Sdim / 64, Sdim / 128), blk, 0, stream>>>(
        qv_h, rp_h, Pbuf, Sdim, Sdim, Dh, Dh, Dh, Sdim);
    // scores = mask((Qu*Kh^T) + shift(P)) / sqrt(D)
    score_kernel<<<dim3(Sdim / 64, Sdim / 128), blk, 0, stream>>>(
        qu_h, k_h, Pbuf, scores);
    // softmax -> f16 probs
    softmax_kernel<<<Sdim, blk, 0, stream>>>(scores, probs);
    // att[:, h*D:(h+1)*D] = probs @ V   (as NT gemm vs V^T)
    gemm_nt_lds<<<dim3(Dh / 64, Sdim / 128), blk, 0, stream>>>(
        probs, vT_h, att32 + (size_t)h * Dh, Sdim, Dh, Sdim, Sdim, Sdim, Vdim);
  }

  // 6) out = att @ Wout^T
  cvt_f32_f16<<<gblk(SV), blk, 0, stream>>>(att32, atth, (int)SV);
  gemm_nt_lds<<<dim3(Vdim / 64, Sdim / 128), blk, 0, stream>>>(
      atth, Wouth, out, Sdim, Vdim, Vdim, Vdim, Vdim, Vdim);
}
